// Regulariser_55035710931684
// MI455X (gfx1250) — matrix operand order repaired and checked by reading, not compiled, not run on hardware
//
#include <hip/hip_runtime.h>
#include <hip/hip_bf16.h>

typedef __attribute__((ext_vector_type(16))) __bf16 v16bf;
typedef __attribute__((ext_vector_type(8)))  __bf16 v8bf;
typedef __attribute__((ext_vector_type(8)))  float  v8f;
typedef __attribute__((ext_vector_type(4)))  int    v4i;

typedef __attribute__((address_space(1))) v4i* global_v4i_p;  // "__device__" AS
typedef __attribute__((address_space(3))) v4i* shared_v4i_p;  // LDS AS

#define N_ROWS 8192
#define DIM    512
#define KT     (DIM / 32)        // 16 k-tiles of depth 32
#define STRIP  32                // columns staged in LDS per iteration
#define LDS_STRIDE (DIM + 16)    // padded row stride (bf16 elems) to break bank conflicts
#define EPSV   1e-6f

// ---- CDNA5 async global->LDS copy (ASYNCcnt path), with safe fallback ----
__device__ __forceinline__ void async_copy_b128(const void* g, void* l) {
#if __has_builtin(__builtin_amdgcn_global_load_async_to_lds_b128)
    __builtin_amdgcn_global_load_async_to_lds_b128(
        (global_v4i_p)(uintptr_t)g,
        (shared_v4i_p)(unsigned)(uintptr_t)l,
        0, 0);
#else
    *(uint4*)l = *(const uint4*)g;   // fallback: bounce through VGPRs
#endif
}

__device__ __forceinline__ void wait_async_done() {
#if __has_builtin(__builtin_amdgcn_s_wait_asynccnt)
    __builtin_amdgcn_s_wait_asynccnt(0);
#else
    asm volatile("s_wait_asynccnt 0x0" ::: "memory");
#endif
}

// ---------------- Kernel 1: fp32 -> bf16 convert ----------------
__global__ void cvt_bf16_kernel(const float* __restrict__ x,
                                __bf16* __restrict__ xb, int n) {
    int i = blockIdx.x * blockDim.x + threadIdx.x;
    if (i < n) xb[i] = (__bf16)x[i];
}

// ---------------- Kernel 2: WMMA gram + fused per-row argmax ----------------
// Block = 256 threads = 8 waves. Wave w owns rows [m0, m0+16), m0 = (blk*8+w)*16.
// A panel (16 rows x 512 K, bf16) resident in registers as 16 WMMA fragments.
// Columns of X staged through LDS in 32-row strips shared by all 8 waves.
__global__ __launch_bounds__(256)
void nn_argmax_kernel(const __bf16* __restrict__ Xb, int* __restrict__ nnIdx) {
    __shared__ __bf16 lds[STRIP * LDS_STRIDE];   // 32 * 528 * 2 = 33 KB

    const int lane = threadIdx.x & 31;
    const int wave = threadIdx.x >> 5;
    const int m0   = (blockIdx.x * 8 + wave) * 16;
    const int half = lane >> 4;                  // 0: lanes 0-15, 1: lanes 16-31
    const int l15  = lane & 15;

    // ---- Load resident A fragments (contiguous k order, same as the B fetch):
    // lane holds row m0+l15; a[0..15] = K = 32t + half*16 + i
    v16bf a[KT];
    {
        const __bf16* arow = Xb + (size_t)(m0 + l15) * DIM;
#pragma unroll
        for (int t = 0; t < KT; ++t) {
            v8bf lo = *(const v8bf*)(arow + 32 * t +     half * 16);
            v8bf hi = *(const v8bf*)(arow + 32 * t + 8 + half * 16);
            a[t] = __builtin_shufflevector(lo, hi,
                       0,1,2,3,4,5,6,7, 8,9,10,11,12,13,14,15);
        }
    }

    // B fragment fetch from LDS (32x16 tile ct, k-tile t):
    // lane holds column ct*16+l15, K = 32t + half*16 + i
    auto ldsB = [&](int ct, int t) -> v16bf {
        return *(const v16bf*)(lds + (ct * 16 + l15) * LDS_STRIDE
                                   + 32 * t + half * 16);
    };

    float bestV[8];
    int   bestI[8];
#pragma unroll
    for (int r = 0; r < 8; ++r) { bestV[r] = -3.4e38f; bestI[r] = 0; }

    const int srow = threadIdx.x >> 3;    // 0..31  (staging: 8 threads/row)
    const int ssub = threadIdx.x & 7;     // 0..7   (each copies 8 x 16B)

    for (int s = 0; s < N_ROWS / STRIP; ++s) {
        __syncthreads();                  // prior strip's LDS reads done
        {   // ---- async-stage strip rows [s*STRIP, s*STRIP+32) into padded LDS
            const uint4* src = (const uint4*)(Xb + (size_t)(s * STRIP + srow) * DIM) + ssub * 8;
            uint4*       dst = (uint4*)(lds + srow * LDS_STRIDE) + ssub * 8;
#pragma unroll
            for (int j = 0; j < 8; ++j) async_copy_b128(src + j, dst + j);
            wait_async_done();            // my transfers landed in LDS
        }
        __syncthreads();                  // everyone's strip data visible

        // ---- 2 column tiles x 16 k-tiles, depth-1 software-pipelined B loads
        v8f   c0 = {}, c1 = {};
        v16bf b0 = ldsB(0, 0), b1 = ldsB(1, 0);
#pragma unroll
        for (int t = 0; t < KT; ++t) {
            v16bf n0 = b0, n1 = b1;
            if (t + 1 < KT) { n0 = ldsB(0, t + 1); n1 = ldsB(1, t + 1); }
            c0 = __builtin_amdgcn_wmma_f32_16x16x32_bf16(
                     false, a[t], false, b0, (short)0, c0, false, false);
            c1 = __builtin_amdgcn_wmma_f32_16x16x32_bf16(
                     false, a[t], false, b1, (short)0, c1, false, false);
            b0 = n0; b1 = n1;
        }

        // ---- argmax update; C layout: c[r] = dots[m0 + half*8 + r][col]
#pragma unroll
        for (int ct = 0; ct < 2; ++ct) {
            const v8f& c = ct ? c1 : c0;
            const int col = s * STRIP + ct * 16 + l15;
#pragma unroll
            for (int r = 0; r < 8; ++r) {
                const int row = m0 + half * 8 + r;
                float v = c[r];
                if (col == row) v = -1.0f;       // mask self-similarity
                if (v > bestV[r]) { bestV[r] = v; bestI[r] = col; }
            }
        }
    }

    // ---- Reduce across the 16 lanes of each half (lower index wins ties,
    // matching argmax's first-max rule; within-lane order was ascending).
#pragma unroll
    for (int off = 8; off; off >>= 1) {
#pragma unroll
        for (int r = 0; r < 8; ++r) {
            float ov = __shfl_xor(bestV[r], off, 32);
            int   oi = __shfl_xor(bestI[r], off, 32);
            if (ov > bestV[r] || (ov == bestV[r] && oi < bestI[r])) {
                bestV[r] = ov; bestI[r] = oi;
            }
        }
    }
    if (l15 == 0) {
#pragma unroll
        for (int r = 0; r < 8; ++r)
            nnIdx[m0 + half * 8 + r] = bestI[r];
    }
}

// ---------------- Kernel 3: exact fp32 per-row log-distance ----------------
__global__ __launch_bounds__(256)
void row_logdist_kernel(const float* __restrict__ x,
                        const int* __restrict__ nnIdx,
                        float* __restrict__ rowLog) {
    const int lane = threadIdx.x & 31;
    const int wave = threadIdx.x >> 5;
    const int row  = blockIdx.x * 8 + wave;
    if (row >= N_ROWS) return;
    const float* xr = x + (size_t)row * DIM;
    const float* xn = x + (size_t)nnIdx[row] * DIM;
    float s = 0.f;
#pragma unroll
    for (int j = 0; j < DIM / 32; ++j) {
        float d = xr[lane + 32 * j] - xn[lane + 32 * j] + EPSV;
        s = fmaf(d, d, s);
    }
#pragma unroll
    for (int off = 16; off; off >>= 1) s += __shfl_xor(s, off, 32);
    if (lane == 0) rowLog[row] = 0.5f * logf(s);   // log(sqrt(s)) = 0.5*log(s)
}

// ---------------- Kernel 4: deterministic final reduction ----------------
__global__ __launch_bounds__(256)
void finalize_kernel(const float* __restrict__ rowLog, float* __restrict__ out) {
    __shared__ float sm[256];
    float s = 0.f;
    for (int i = threadIdx.x; i < N_ROWS; i += 256) s += rowLog[i];
    sm[threadIdx.x] = s;
    __syncthreads();
    for (int off = 128; off; off >>= 1) {
        if ((int)threadIdx.x < off) sm[threadIdx.x] += sm[threadIdx.x + off];
        __syncthreads();
    }
    if (threadIdx.x == 0) out[0] = -sm[0] / (float)N_ROWS;
}

extern "C" void kernel_launch(void* const* d_in, const int* in_sizes, int n_in,
                              void* d_out, int out_size, void* d_ws, size_t ws_size,
                              hipStream_t stream) {
    const float* x = (const float*)d_in[0];

    // workspace layout: [bf16 X : 8 MB][nn indices : 32 KB][row logs : 32 KB]
    __bf16* xb     = (__bf16*)d_ws;
    char*   base   = (char*)d_ws + (size_t)N_ROWS * DIM * sizeof(__bf16);
    int*    nnIdx  = (int*)base;
    float*  rowLog = (float*)(base + N_ROWS * sizeof(int));
    float*  out    = (float*)d_out;

    cvt_bf16_kernel<<<(N_ROWS * DIM) / 256, 256, 0, stream>>>(x, xb, N_ROWS * DIM);
    nn_argmax_kernel<<<N_ROWS / 128, 256, 0, stream>>>(xb, nnIdx);
    row_logdist_kernel<<<N_ROWS / 8, 256, 0, stream>>>(x, nnIdx, rowLog);
    finalize_kernel<<<1, 256, 0, stream>>>(rowLog, out);
}
